// HFNSACore_12859132084900
// MI455X (gfx1250) — matrix operand order repaired and checked by reading, not compiled, not run on hardware
//
#include <hip/hip_runtime.h>
#include <hip/hip_bf16.h>

// ---- NSA constants (match reference) ----
#define T_SEQ   2048
#define HQ      16
#define DD      128
#define KS_     32
#define ST_     16
#define BS_     64
#define TOPN    16
#define NINIT_  2
#define NLOCAL_ 1
#define WIN_    512
#define MM_     127     // number of compressed tokens
#define MPAD    128
#define NBK     32      // number of selection blocks
#define NEGF    (-1e30f)
#define BIGF    (1e30f)

typedef __attribute__((ext_vector_type(16))) _Float16 v16h;
typedef __attribute__((ext_vector_type(8)))  _Float16 v8h;
typedef __attribute__((ext_vector_type(8)))  float    v8f;
typedef __attribute__((ext_vector_type(4)))  int      vi4;

// ---- CDNA5 async global->LDS path (guarded; falls back to VGPR staging) ----
#if defined(__has_builtin)
#  if __has_builtin(__builtin_amdgcn_global_load_async_to_lds_b128) && \
      __has_builtin(__builtin_amdgcn_s_wait_asynccnt)
#    define HAVE_ASYNC_LDS 1
#  endif
#endif
#ifndef HAVE_ASYNC_LDS
#  define HAVE_ASYNC_LDS 0
#endif

// Exact parameter types per the clang diagnostic:
//   param0: int4 addrspace(1)* (printed '__device__'), param1: int4 addrspace(3)*
typedef __attribute__((address_space(1))) vi4* gvi4_p;
typedef __attribute__((address_space(3))) vi4* lvi4_p;

// Stage 16 bytes (8 f16) from global to LDS, one lane's worth.
// LDS pointer goes through an integer cast: generic pointers into the LDS
// aperture carry the LDS byte offset in addr[31:0] (ISA 10.2).
__device__ __forceinline__ void stage16(const _Float16* gsrc, _Float16* ldst) {
#if HAVE_ASYNC_LDS
  __builtin_amdgcn_global_load_async_to_lds_b128(
      (gvi4_p)(unsigned long long)gsrc,
      (lvi4_p)(unsigned)(unsigned long long)ldst, 0, 0);
#else
  *(v8h*)ldst = *(const v8h*)gsrc;
#endif
}
__device__ __forceinline__ void stage_join() {
#if HAVE_ASYNC_LDS
  __builtin_amdgcn_s_wait_asynccnt(0);
#endif
  __syncthreads();
}

__device__ __forceinline__ v8f wmma_fp16(v16h a, v16h b, v8f c) {
  // D = A(16x32 f16) * B(32x16 f16) + C(16x16 f32)
  return __builtin_amdgcn_wmma_f32_16x16x32_f16(false, a, false, b, (short)0, c, false, false);
}

__device__ __forceinline__ v8f zero8() {
  v8f z;
#pragma unroll
  for (int i = 0; i < 8; ++i) z[i] = 0.f;
  return z;
}

// A-fragment (16x32, f16) from row-major f16 row; lane half hh selects K sub-ranges:
// elems 0..7  <- row[hh*16 + 0..7], elems 8..15 <- row[hh*16 + 8..15]
__device__ __forceinline__ v16h load_afrag_h(const _Float16* row, int hh) {
  v8h lo = *(const v8h*)(row + hh * 16);
  v8h hi = *(const v8h*)(row + hh * 16 + 8);
  v16h r;
#pragma unroll
  for (int i = 0; i < 8; ++i) { r[i] = lo[i]; r[i + 8] = hi[i]; }
  return r;
}

// Row-wise reductions across the 16 lanes holding one C-layout row
__device__ __forceinline__ float redmax16(float x) {
#pragma unroll
  for (int m = 1; m <= 8; m <<= 1) x = fmaxf(x, __shfl_xor(x, m, 32));
  return x;
}
__device__ __forceinline__ float redsum16(float x) {
#pragma unroll
  for (int m = 1; m <= 8; m <<= 1) x += __shfl_xor(x, m, 32);
  return x;
}
__device__ __forceinline__ float sigmoidf_(float x) { return 1.0f / (1.0f + __expf(-x)); }

// ---------------- prep kernels ----------------
__global__ void prep_convert(const float* __restrict__ k, const float* __restrict__ v,
                             _Float16* __restrict__ kh, _Float16* __restrict__ vT) {
  const int t = blockIdx.x, d = threadIdx.x;
  kh[t * DD + d]    = (_Float16)k[t * DD + d];
  vT[d * T_SEQ + t] = (_Float16)v[t * DD + d];
}

__global__ void prep_q(const float* __restrict__ q, _Float16* __restrict__ qh) {
  const size_t i = (size_t)blockIdx.x * DD + threadIdx.x;
  qh[i] = (_Float16)q[i];
}

__global__ void prep_pool(const float* __restrict__ k, const float* __restrict__ v,
                          _Float16* __restrict__ ckh, _Float16* __restrict__ cvT) {
  const int m = blockIdx.x, d = threadIdx.x;
  float sk = 0.f, sv = 0.f;
  if (m < MM_) {
#pragma unroll 4
    for (int i = 0; i < KS_; ++i) {
      sk += k[(m * ST_ + i) * DD + d];
      sv += v[(m * ST_ + i) * DD + d];
    }
    sk *= (1.0f / KS_); sv *= (1.0f / KS_);
  }
  ckh[m * DD + d]   = (_Float16)sk;   // padded row 127 = 0
  cvT[d * MPAD + m] = (_Float16)sv;
}

// ---------------- compressed attention + topk ----------------
__global__ void __launch_bounds__(512) cmp_attn_kernel(
    const _Float16* __restrict__ qh, const float* __restrict__ cw,
    const _Float16* __restrict__ ckh, const _Float16* __restrict__ cvT,
    float* __restrict__ out, int* __restrict__ topk) {
  __shared__ float    slcp[16][NBK];        // block scores, summed over heads
  __shared__ _Float16 pst[16][16][32];      // per-wave P staging (C-layout -> A-layout)

  const int qbase = blockIdx.x * 16;
  const int wv    = threadIdx.x >> 5;       // wave index == q head
  const int lane  = threadIdx.x & 31;
  const int n     = lane & 15;
  const int hh    = lane >> 4;
  const float scale = 0.08838834764831845f; // 1/sqrt(128)

  slcp[threadIdx.x >> 5][threadIdx.x & 31] = 0.f;  // 512 threads cover 16*32
  __syncthreads();

  // Q A-fragments, 4 chunks of K=32 over D=128
  v16h aq[4];
#pragma unroll
  for (int c = 0; c < 4; ++c)
    aq[c] = load_afrag_h(qh + ((size_t)(qbase + n) * HQ + wv) * DD + c * 32, hh);

  // scores over 8 key tiles (M=127 padded to 128)
  v8f s[8];
#pragma unroll
  for (int jt = 0; jt < 8; ++jt) {
    v8f a = zero8();
#pragma unroll
    for (int c = 0; c < 4; ++c) {
      v16h bk = *(const v16h*)(ckh + (size_t)(jt * 16 + n) * DD + c * 32 + hh * 16);
      a = wmma_fp16(aq[c], bk, a);
    }
    s[jt] = a;
  }

  // softmax per row (row m lives on fixed VGPR r + lane-half; cols across 16 lanes)
  float w0[8];
#pragma unroll
  for (int r = 0; r < 8; ++r) {
    const int m = r + 8 * hh;
    const int t = qbase + m;
    float mx = -INFINITY;
#pragma unroll
    for (int jt = 0; jt < 8; ++jt) {
      const int j = jt * 16 + n;
      const bool valid = (j < MM_) && (j * ST_ + KS_ <= t + 1);
      const float sv = valid ? s[jt][r] * scale : -INFINITY;
      s[jt][r] = sv;
      mx = fmaxf(mx, sv);
    }
    mx = redmax16(mx);
    float sum = 0.f;
#pragma unroll
    for (int jt = 0; jt < 8; ++jt) {
      const float e = (s[jt][r] == -INFINITY) ? 0.f : __expf(s[jt][r] - mx);
      s[jt][r] = e;
      sum += e;
    }
    sum = redsum16(sum);
    const float inv = (sum > 0.f) ? 1.0f / sum : 0.f;   // rows with no valid keys -> p = 0
#pragma unroll
    for (int jt = 0; jt < 8; ++jt) s[jt][r] *= inv;     // probabilities now
    w0[r] = sigmoidf_(cw[((size_t)t * HQ + wv) * 3 + 0]);
  }

  // slc_p: sparse Wmap (token j -> block j>>2, split 0.5/0.5 when j%4==3), sum over heads
#pragma unroll
  for (int r = 0; r < 8; ++r) {
    const int m = r + 8 * hh;
#pragma unroll
    for (int jt = 0; jt < 8; ++jt) {
      const int j = jt * 16 + n;
      const float p = s[jt][r];
      if (j < MM_ && p > 0.f) {
        const int b = j >> 2;
        if ((j & 3) == 3) {
          atomicAdd(&slcp[m][b], 0.5f * p);
          atomicAdd(&slcp[m][b + 1], 0.5f * p);
        } else {
          atomicAdd(&slcp[m][b], p);
        }
      }
    }
  }

  // P: C-layout -> A-layout via LDS (per-wave buffer), 4 chunks of 32 keys
  v16h pa[4];
#pragma unroll
  for (int kc = 0; kc < 4; ++kc) {
    __syncthreads();
#pragma unroll
    for (int r = 0; r < 8; ++r) {
      const int m = r + 8 * hh;
      pst[wv][m][n]      = (_Float16)s[kc * 2 + 0][r];
      pst[wv][m][16 + n] = (_Float16)s[kc * 2 + 1][r];
    }
    __syncthreads();
    pa[kc] = load_afrag_h(&pst[wv][n][0], hh);
  }

  // cmp_o = P @ cmp_V, write out = sigmoid(w0) * cmp_o  (full overwrite)
#pragma unroll
  for (int dt = 0; dt < 8; ++dt) {
    v8f a = zero8();
#pragma unroll
    for (int kc = 0; kc < 4; ++kc) {
      v16h bv = *(const v16h*)(cvT + (size_t)(dt * 16 + n) * MPAD + kc * 32 + hh * 16);
      a = wmma_fp16(pa[kc], bv, a);
    }
#pragma unroll
    for (int r = 0; r < 8; ++r) {
      const int t = qbase + r + 8 * hh;
      out[((size_t)t * HQ + wv) * DD + dt * 16 + n] = w0[r] * a[r];
    }
  }

  // top-16 block selection per query row
  __syncthreads();
  if (threadIdx.x < 16) {
    const int m = threadIdx.x;
    const int t = qbase + m;
    const int cur = t / BS_;
    float vals[NBK];
#pragma unroll
    for (int b = 0; b < NBK; ++b) {
      float x = slcp[m][b];
      if (b > cur) x = NEGF;                     // future blocks
      if (b < NINIT_ || b == cur) x = BIGF;      // forced blocks (NLOCAL=1)
      vals[b] = x;
    }
#pragma unroll 1
    for (int i = 0; i < TOPN; ++i) {
      int best = 0; float bv = vals[0];
      for (int b = 1; b < NBK; ++b) if (vals[b] > bv) { bv = vals[b]; best = b; }
      topk[t * TOPN + i] = best;
      vals[best] = -INFINITY;
    }
  }
}

// ---------------- flash core for selection / sliding-window ----------------
// K/V tiles for the 32-key chunk are staged once per workgroup into LDS
// (async global->LDS on CDNA5), then all 16 head-waves consume them.
template <int MODE>  // 0 = selection, 1 = sliding window
__device__ __forceinline__ void process_chunk(
    int key0, int blk, int qbase, int n, int hh, int tid,
    const _Float16* __restrict__ kh, const _Float16* __restrict__ vT,
    _Float16* klds, _Float16* vlds,
    const unsigned* rowmask, _Float16 (*pw)[32],
    v16h aq[4], v8f acc[8], float rmax[8], float rsum[8]) {
  const float scale = 0.08838834764831845f;

  // ---- stage K tile [32 keys][128 d] and V tile [128 dv][32 keys] into LDS ----
  __syncthreads();  // WAR: previous chunk's LDS consumers are done
  stage16(kh + (size_t)key0 * DD + tid * 8, klds + tid * 8);
  {
    const int dv = tid >> 2, c4 = tid & 3;
    stage16(vT + (size_t)dv * T_SEQ + key0 + c4 * 8, vlds + dv * 32 + c4 * 8);
  }
  stage_join();     // s_wait_asynccnt 0 + barrier

  // ---- scores: two 16-key tiles ----
  v8f s0 = zero8(), s1 = zero8();
#pragma unroll
  for (int c = 0; c < 4; ++c) {
    v16h b0 = *(const v16h*)(klds + (size_t)n * DD + c * 32 + hh * 16);
    v16h b1 = *(const v16h*)(klds + (size_t)(16 + n) * DD + c * 32 + hh * 16);
    s0 = wmma_fp16(aq[c], b0, s0);
    s1 = wmma_fp16(aq[c], b1, s1);
  }
  float alpha[8];
#pragma unroll
  for (int r = 0; r < 8; ++r) {
    const int m = r + 8 * hh;
    const int t = qbase + m;
    const int sa = key0 + n, sb = key0 + 16 + n;
    bool va, vb;
    if (MODE == 0) {
      const bool mem = (rowmask[m] >> blk) & 1u;
      va = mem && (sa <= t);
      vb = mem && (sb <= t);
    } else {
      va = (sa <= t) && (t - sa < WIN_);
      vb = (sb <= t) && (t - sb < WIN_);
    }
    const float x0 = va ? s0[r] * scale : -INFINITY;
    const float x1 = vb ? s1[r] * scale : -INFINITY;
    float mx = redmax16(fmaxf(x0, x1));
    const float nm = fmaxf(rmax[r], mx);
    const float al = (rmax[r] == -INFINITY) ? 0.f : __expf(rmax[r] - nm);
    const float e0 = (x0 == -INFINITY) ? 0.f : __expf(x0 - nm);
    const float e1 = (x1 == -INFINITY) ? 0.f : __expf(x1 - nm);
    rsum[r] = rsum[r] * al + redsum16(e0 + e1);
    rmax[r] = nm;
    alpha[r] = al;
    s0[r] = e0; s1[r] = e1;
  }
#pragma unroll
  for (int dt = 0; dt < 8; ++dt)
#pragma unroll
    for (int r = 0; r < 8; ++r) acc[dt][r] *= alpha[r];

  // ---- P: C-layout -> A-layout via per-wave LDS buffer ----
#pragma unroll
  for (int r = 0; r < 8; ++r) {
    const int m = r + 8 * hh;
    pw[m][n]      = (_Float16)s0[r];
    pw[m][16 + n] = (_Float16)s1[r];
  }
  __syncthreads();
  const v16h pa = load_afrag_h(&pw[n][0], hh);
#pragma unroll
  for (int dt = 0; dt < 8; ++dt) {
    v16h bv = *(const v16h*)(vlds + (size_t)(dt * 16 + n) * 32 + hh * 16);
    acc[dt] = wmma_fp16(pa, bv, acc[dt]);
  }
}

template <int MODE>
__global__ void __launch_bounds__(512) sparse_attn_kernel(
    const _Float16* __restrict__ qh, const float* __restrict__ cw,
    const _Float16* __restrict__ kh, const _Float16* __restrict__ vT,
    const int* __restrict__ topk, float* __restrict__ out) {
  __shared__ _Float16 pst[16][16][32];   // 16 KB: per-wave P transpose staging
  __shared__ _Float16 klds[32 * DD];     //  8 KB: K tile, shared by all 16 head-waves
  __shared__ _Float16 vlds[DD * 32];     //  8 KB: V tile (transposed), shared
  __shared__ unsigned rowmask[16];
  __shared__ unsigned umask_sh;

  const int qbase = blockIdx.x * 16;
  const int tid   = threadIdx.x;
  const int wv    = tid >> 5;
  const int lane  = tid & 31;
  const int n     = lane & 15;
  const int hh    = lane >> 4;

  if (MODE == 0) {
    if (tid == 0) umask_sh = 0u;
    __syncthreads();
    if (tid < 16) {
      const int t = qbase + tid;
      unsigned msk = 0u;
      for (int i = 0; i < TOPN; ++i) {
        const int b = topk[t * TOPN + i];
        if (b * BS_ <= t) msk |= (1u << b);   // drop fully-future selections (causal kills them)
      }
      rowmask[tid] = msk;
      atomicOr(&umask_sh, msk);
    }
    __syncthreads();
  }

  v16h aq[4];
#pragma unroll
  for (int c = 0; c < 4; ++c)
    aq[c] = load_afrag_h(qh + ((size_t)(qbase + n) * HQ + wv) * DD + c * 32, hh);

  v8f acc[8];
  float rmax[8], rsum[8];
#pragma unroll
  for (int r = 0; r < 8; ++r) { acc[r] = zero8(); rmax[r] = -INFINITY; rsum[r] = 0.f; }

  if (MODE == 0) {
    unsigned um = umask_sh;          // uniform across workgroup
    while (um) {
      const int b = __builtin_ctz(um);
      um &= (um - 1);
#pragma unroll 1
      for (int kc2 = 0; kc2 < 2; ++kc2)
        process_chunk<0>(b * BS_ + kc2 * 32, b, qbase, n, hh, tid, kh, vT,
                         klds, vlds, rowmask, &pst[wv][0], aq, acc, rmax, rsum);
    }
  } else {
    const int c0 = (qbase >= (WIN_ - 1)) ? ((qbase - (WIN_ - 1)) >> 5) : 0;
    const int c1 = (qbase + 15) >> 5;
#pragma unroll 1
    for (int c = c0; c <= c1; ++c) {
      if (c < c1)
        __builtin_prefetch((const void*)(kh + (size_t)(c + 1) * 32 * DD), 0, 1);
      process_chunk<1>(c * 32, 0, qbase, n, hh, tid, kh, vT,
                       klds, vlds, rowmask, &pst[wv][0], aq, acc, rmax, rsum);
    }
  }

  float wsc[8];
#pragma unroll
  for (int r = 0; r < 8; ++r) {
    const int t = qbase + r + 8 * hh;
    const float inv = (rsum[r] > 0.f) ? 1.0f / rsum[r] : 0.f;
    wsc[r] = sigmoidf_(cw[((size_t)t * HQ + wv) * 3 + (MODE == 0 ? 1 : 2)]) * inv;
  }
#pragma unroll
  for (int dt = 0; dt < 8; ++dt) {
#pragma unroll
    for (int r = 0; r < 8; ++r) {
      const int t = qbase + r + 8 * hh;
      const size_t idx = ((size_t)t * HQ + wv) * DD + dt * 16 + n;
      out[idx] += wsc[r] * acc[dt][r];
    }
  }
}

// ---------------- host side ----------------
extern "C" void kernel_launch(void* const* d_in, const int* in_sizes, int n_in,
                              void* d_out, int out_size, void* d_ws, size_t ws_size,
                              hipStream_t stream) {
  (void)in_sizes; (void)n_in; (void)out_size; (void)ws_size;
  const float* q  = (const float*)d_in[0];
  const float* k  = (const float*)d_in[1];
  const float* v  = (const float*)d_in[2];
  const float* cw = (const float*)d_in[3];
  // d_in[4] = cu_seqlens [0, T] — single sequence, unused
  float* out = (float*)d_out;

  _Float16* kh  = (_Float16*)d_ws;                  // [T][128] f16 row-major
  _Float16* vT  = kh  + (size_t)T_SEQ * DD;         // [128][T] f16 transposed
  _Float16* ckh = vT  + (size_t)DD * T_SEQ;         // [128][128] cmp K (row 127 zero)
  _Float16* cvT = ckh + (size_t)MPAD * DD;          // [128][128] cmp V transposed
  _Float16* qh  = cvT + (size_t)DD * MPAD;          // [T][16][128] f16 Q
  int* topk     = (int*)(qh + (size_t)T_SEQ * HQ * DD);  // [T][16]

  prep_convert<<<T_SEQ, DD, 0, stream>>>(k, v, kh, vT);
  prep_q<<<T_SEQ * HQ, DD, 0, stream>>>(q, qh);
  prep_pool<<<MPAD, DD, 0, stream>>>(k, v, ckh, cvT);
  cmp_attn_kernel<<<T_SEQ / 16, 512, 0, stream>>>(qh, cw, ckh, cvT, out, topk);
  sparse_attn_kernel<0><<<T_SEQ / 16, 512, 0, stream>>>(qh, cw, kh, vT, topk, out);
  sparse_attn_kernel<1><<<T_SEQ / 16, 512, 0, stream>>>(qh, cw, kh, vT, topk, out);
}
